// GCN_29824252903679
// MI455X (gfx1250) — matrix operand order repaired and checked by reading, not compiled, not run on hardware
//
#include <hip/hip_runtime.h>

typedef __attribute__((ext_vector_type(16))) __bf16   v16bf;
typedef __attribute__((ext_vector_type(8)))  float    v8f;
typedef __attribute__((ext_vector_type(4)))  float    v4f;
typedef __attribute__((ext_vector_type(8)))  unsigned v8u;

#define NFEAT  512
#define NHID   256
#define NCLASS 40
#define NCPAD  48    // NCLASS padded to 3x16 tiles

// ---------------------------------------------------------------------------
// Pack a [K x Csrc] f32 weight matrix into split-bf16 hi/lo arrays laid out in
// WMMA B-operand register order:
//   dword index = ((ktIdx*(Cpad/16) + tc)*32 + lane)*8 + v
// so a consuming lane reads its whole 16-bf16 B fragment as one 32-byte load.
// Columns >= Csrc are zero-padded (handles NCLASS=40 -> 48).
// ---------------------------------------------------------------------------
__global__ void gcn_pack_w(const float* __restrict__ W,
                           unsigned* __restrict__ hi, unsigned* __restrict__ lo,
                           int K, int Cpad, int Csrc) {
    const unsigned gid = blockIdx.x * blockDim.x + threadIdx.x;
    const unsigned total = (unsigned)(K / 2) * (unsigned)Cpad;
    if (gid >= total) return;
    const unsigned v    = gid & 7u;
    const unsigned lane = (gid >> 3) & 31u;
    const unsigned tile = gid >> 8;
    const unsigned ntc  = (unsigned)Cpad / 16u;
    const unsigned tc   = tile % ntc;
    const unsigned ktI  = tile / ntc;
    const int half = (int)(lane >> 4);
    const int c = (int)(tc * 16u + (lane & 15u));
    const int k = (int)(ktI * 32u) + 2 * (int)(v & 3u) + 8 * half + 16 * (int)(v >> 2);  // B 32x16 layout
    float b0 = 0.f, b1 = 0.f;
    if (c < Csrc) {
        b0 = W[(size_t)k * Csrc + c];
        b1 = W[(size_t)(k + 1) * Csrc + c];
    }
    const __bf16 g0 = (__bf16)b0, g1 = (__bf16)b1;
    const __bf16 e0 = (__bf16)(b0 - (float)g0), e1 = (__bf16)(b1 - (float)g1);
    const unsigned h0 = __builtin_bit_cast(unsigned short, g0);
    const unsigned h1 = __builtin_bit_cast(unsigned short, g1);
    const unsigned l0 = __builtin_bit_cast(unsigned short, e0);
    const unsigned l1 = __builtin_bit_cast(unsigned short, e1);
    hi[gid] = h0 | (h1 << 16);
    lo[gid] = l0 | (l1 << 16);
}

// ---------------------------------------------------------------------------
// GEMM1: sup1 = x @ W1   (fp32 via 3x bf16 WMMA: hi*hi + lo*hi + hi*lo)
// One wave32 computes 16x64 (4 N-tiles): A loaded/split once per K-tile as
// 4x b128 NT loads; each B tile fragment is 2x 32B loads of pre-packed bf16.
// ---------------------------------------------------------------------------
__global__ void gcn_gemm1(const float* __restrict__ x,
                          const unsigned* __restrict__ Whi,
                          const unsigned* __restrict__ Wlo,
                          float* __restrict__ out, int nrows) {
    const int lane = threadIdx.x & 31;
    const int wave = (blockIdx.x * blockDim.x + threadIdx.x) >> 5;
    const int ngroups = NHID / 64;                 // 4 groups of 4 N-tiles
    const int nwaves  = (nrows / 16) * ngroups;
    if (wave >= nwaves) return;                    // wave-uniform
    const int tg = wave % ngroups;
    const int tm = wave / ngroups;
    const int half = lane >> 4;
    const int l    = lane & 15;
    const int arow = tm * 16 + l;                  // A: M = lane&15 (both halves)
    const int ntc  = NHID / 16;                    // 16 col tiles total

    v8f acc0 = {0.f,0.f,0.f,0.f,0.f,0.f,0.f,0.f};
    v8f acc1 = acc0, acc2 = acc0, acc3 = acc0;

    const float* ap = x + (size_t)arow * NFEAT;
    for (int ktI = 0; ktI < NFEAT / 32; ++ktI) {
        const int kb = ktI * 32 + 8 * half;
        // A fragment = two contiguous 8-float runs -> 4x b128 NT loads
        const v4f a0 = __builtin_nontemporal_load((const v4f*)(ap + kb));
        const v4f a1 = __builtin_nontemporal_load((const v4f*)(ap + kb + 4));
        const v4f a2 = __builtin_nontemporal_load((const v4f*)(ap + kb + 16));
        const v4f a3 = __builtin_nontemporal_load((const v4f*)(ap + kb + 20));
        v16bf ahi, alo;
#pragma unroll
        for (int j = 0; j < 4; ++j) {
            __bf16 h;
            h = (__bf16)a0[j]; ahi[j]      = h; alo[j]      = (__bf16)(a0[j] - (float)h);
            h = (__bf16)a1[j]; ahi[4 + j]  = h; alo[4 + j]  = (__bf16)(a1[j] - (float)h);
            h = (__bf16)a2[j]; ahi[8 + j]  = h; alo[8 + j]  = (__bf16)(a2[j] - (float)h);
            h = (__bf16)a3[j]; ahi[12 + j] = h; alo[12 + j] = (__bf16)(a3[j] - (float)h);
        }
        auto dotile = [&](int jb, v8f& accr) {
            const size_t base = ((size_t)(ktI * ntc + tg * 4 + jb) * 32 + lane) * 8;
            const v16bf bhi = __builtin_bit_cast(v16bf, *(const v8u*)(Whi + base));
            const v16bf blo = __builtin_bit_cast(v16bf, *(const v8u*)(Wlo + base));
            accr = __builtin_amdgcn_wmma_f32_16x16x32_bf16(false, ahi, false, bhi, (short)0, accr, false, false);
            accr = __builtin_amdgcn_wmma_f32_16x16x32_bf16(false, alo, false, bhi, (short)0, accr, false, false);
            accr = __builtin_amdgcn_wmma_f32_16x16x32_bf16(false, ahi, false, blo, (short)0, accr, false, false);
        };
        dotile(0, acc0); dotile(1, acc1); dotile(2, acc2); dotile(3, acc3);
    }

    auto store = [&](int jb, const v8f& accr) {
#pragma unroll
        for (int r = 0; r < 8; ++r) {
            const int m = tm * 16 + r + 8 * half;  // C/D layout
            out[(size_t)m * NHID + tg * 64 + jb * 16 + l] = accr[r];
        }
    };
    store(0, acc0); store(1, acc1); store(2, acc2); store(3, acc3);
}

// ---------------------------------------------------------------------------
// GEMM2: sup2 = relu(h1) @ W2   (3 padded N-tiles per wave, fused ReLU,
// pre-packed zero-padded B; h1 is read for the last time -> NT loads)
// ---------------------------------------------------------------------------
__global__ void gcn_gemm2(const float* __restrict__ h,
                          const unsigned* __restrict__ Whi,
                          const unsigned* __restrict__ Wlo,
                          float* __restrict__ out, int nrows) {
    const int lane = threadIdx.x & 31;
    const int wave = (blockIdx.x * blockDim.x + threadIdx.x) >> 5;
    const int nwaves = nrows / 16;
    if (wave >= nwaves) return;                    // wave-uniform
    const int tm = wave;
    const int half = lane >> 4;
    const int l    = lane & 15;
    const int arow = tm * 16 + l;
    const int ntc  = NCPAD / 16;                   // 3

    v8f acc0 = {0.f,0.f,0.f,0.f,0.f,0.f,0.f,0.f};
    v8f acc1 = acc0, acc2 = acc0;

    const float* ap = h + (size_t)arow * NHID;
    for (int ktI = 0; ktI < NHID / 32; ++ktI) {
        const int kb = ktI * 32 + 8 * half;
        v4f a0 = __builtin_nontemporal_load((const v4f*)(ap + kb));
        v4f a1 = __builtin_nontemporal_load((const v4f*)(ap + kb + 4));
        v4f a2 = __builtin_nontemporal_load((const v4f*)(ap + kb + 16));
        v4f a3 = __builtin_nontemporal_load((const v4f*)(ap + kb + 20));
        v16bf ahi, alo;
#pragma unroll
        for (int j = 0; j < 4; ++j) {
            float f; __bf16 hh;
            f = fmaxf(a0[j], 0.f); hh = (__bf16)f; ahi[j]      = hh; alo[j]      = (__bf16)(f - (float)hh);
            f = fmaxf(a1[j], 0.f); hh = (__bf16)f; ahi[4 + j]  = hh; alo[4 + j]  = (__bf16)(f - (float)hh);
            f = fmaxf(a2[j], 0.f); hh = (__bf16)f; ahi[8 + j]  = hh; alo[8 + j]  = (__bf16)(f - (float)hh);
            f = fmaxf(a3[j], 0.f); hh = (__bf16)f; ahi[12 + j] = hh; alo[12 + j] = (__bf16)(f - (float)hh);
        }
        auto dotile = [&](int jb, v8f& accr) {
            const size_t base = ((size_t)(ktI * ntc + jb) * 32 + lane) * 8;
            const v16bf bhi = __builtin_bit_cast(v16bf, *(const v8u*)(Whi + base));
            const v16bf blo = __builtin_bit_cast(v16bf, *(const v8u*)(Wlo + base));
            accr = __builtin_amdgcn_wmma_f32_16x16x32_bf16(false, ahi, false, bhi, (short)0, accr, false, false);
            accr = __builtin_amdgcn_wmma_f32_16x16x32_bf16(false, alo, false, bhi, (short)0, accr, false, false);
            accr = __builtin_amdgcn_wmma_f32_16x16x32_bf16(false, ahi, false, blo, (short)0, accr, false, false);
        };
        dotile(0, acc0); dotile(1, acc1); dotile(2, acc2);
    }

    auto store = [&](int jb, const v8f& accr) {
        const int c = jb * 16 + l;
        if (c < NCLASS) {
#pragma unroll
            for (int r = 0; r < 8; ++r) {
                const int m = tm * 16 + r + 8 * half;
                out[(size_t)m * NCLASS + c] = accr[r];
            }
        }
    };
    store(0, acc0); store(1, acc1); store(2, acc2);
}

// ---------------------------------------------------------------------------
// Initialize an [nrows, ncols] accumulator with a broadcast bias row
// ---------------------------------------------------------------------------
__global__ void gcn_init_bias(float* __restrict__ out, const float* __restrict__ b,
                              int nrows, int ncols) {
    const unsigned gid = blockIdx.x * blockDim.x + threadIdx.x;
    if (gid >= (unsigned)(nrows * ncols)) return;
    out[gid] = b[gid % (unsigned)ncols];
}

// ---------------------------------------------------------------------------
// Scatter layer 1: h1[dst[e]] += ew[e] * sup1[src[e]]   (256 feats, float4/thr)
// sup1 and h1 (102 MB each) live in the 192 MB L2; edge streams (read once)
// use non-temporal loads so they don't evict them.
// ---------------------------------------------------------------------------
__global__ void gcn_scatter1(const float* __restrict__ sup,
                             const int* __restrict__ src,
                             const int* __restrict__ dst,
                             const float* __restrict__ ew,
                             float* __restrict__ out, int nedges) {
    const unsigned gid = blockIdx.x * blockDim.x + threadIdx.x;
    const unsigned e = gid >> 6;                   // 64 threads per edge
    if (e >= (unsigned)nedges) return;
    const int f = (int)(gid & 63u) * 4;
    const int s = __builtin_nontemporal_load(src + e);
    const int d = __builtin_nontemporal_load(dst + e);
    const float w = __builtin_nontemporal_load(ew + e);
    const float4 m = *(const float4*)(sup + (size_t)s * NHID + f);
    float* o = out + (size_t)d * NHID + f;
    atomicAdd(o + 0, w * m.x);
    atomicAdd(o + 1, w * m.y);
    atomicAdd(o + 2, w * m.z);
    atomicAdd(o + 3, w * m.w);
}

// ---------------------------------------------------------------------------
// Scatter layer 2: 40 features, one wave per edge (lane + lane+32 for lane<8)
// ---------------------------------------------------------------------------
__global__ void gcn_scatter2(const float* __restrict__ sup,
                             const int* __restrict__ src,
                             const int* __restrict__ dst,
                             const float* __restrict__ ew,
                             float* __restrict__ out, int nedges) {
    const unsigned gid = blockIdx.x * blockDim.x + threadIdx.x;
    const unsigned e = gid >> 5;
    const int lane = (int)(gid & 31u);
    if (e >= (unsigned)nedges) return;
    const int s = __builtin_nontemporal_load(src + e);
    const int d = __builtin_nontemporal_load(dst + e);
    const float w = __builtin_nontemporal_load(ew + e);
    atomicAdd(out + (size_t)d * NCLASS + lane, w * sup[(size_t)s * NCLASS + lane]);
    if (lane < 8)
        atomicAdd(out + (size_t)d * NCLASS + 32 + lane,
                  w * sup[(size_t)s * NCLASS + 32 + lane]);
}

// ---------------------------------------------------------------------------
// Row-wise log-softmax over 40 classes, one wave32 per row (last read -> NT)
// ---------------------------------------------------------------------------
__global__ void gcn_logsoftmax(const float* __restrict__ a,
                               float* __restrict__ out, int nrows) {
    const unsigned gid = blockIdx.x * blockDim.x + threadIdx.x;
    const unsigned row = gid >> 5;
    const int lane = (int)(gid & 31u);
    if (row >= (unsigned)nrows) return;
    const float* r = a + (size_t)row * NCLASS;
    const float v0 = __builtin_nontemporal_load(r + lane);
    const float v1 = (lane < 8) ? __builtin_nontemporal_load(r + 32 + lane) : -3.0e38f;
    float m = fmaxf(v0, v1);
    for (int off = 16; off > 0; off >>= 1) m = fmaxf(m, __shfl_xor(m, off, 32));
    float s = expf(v0 - m) + ((lane < 8) ? expf(v1 - m) : 0.f);
    for (int off = 16; off > 0; off >>= 1) s += __shfl_xor(s, off, 32);
    const float lse = m + logf(s);
    float* o = out + (size_t)row * NCLASS;
    o[lane] = v0 - lse;
    if (lane < 8) o[32 + lane] = v1 - lse;
}

// ---------------------------------------------------------------------------
extern "C" void kernel_launch(void* const* d_in, const int* in_sizes, int n_in,
                              void* d_out, int out_size, void* d_ws, size_t ws_size,
                              hipStream_t stream) {
    const float* x   = (const float*)d_in[0];
    const int*  eidx = (const int*) d_in[1];
    const float* ew  = (const float*)d_in[2];
    const float* W1  = (const float*)d_in[3];
    const float* b1  = (const float*)d_in[4];
    const float* W2  = (const float*)d_in[5];
    const float* b2  = (const float*)d_in[6];
    float* out = (float*)d_out;

    const int N = in_sizes[0] / NFEAT;   // 100000
    const int E = in_sizes[2];           // 3200000
    const int* src = eidx;               // edge_index[0]
    const int* dst = eidx + E;           // edge_index[1]

    char* ws = (char*)d_ws;
    float* sup1 = (float*)ws;                                 // [N, 256]
    float* h1   = (float*)(ws + (size_t)N * NHID * 4);        // [N, 256]
    float* sup2 = (float*)ws;                                 // [N, 40]  (reuses sup1)
    float* agg2 = (float*)(ws + (size_t)N * NCLASS * 4);      // [N, 40]  (reuses sup1)
    // packed weights live past h1 (W1: 2x256KB, W2: 2x24KB)
    const size_t PK = (size_t)2 * N * NHID * 4;
    const size_t w1n = (size_t)(NFEAT / 2) * NHID;            // dwords
    const size_t w2n = (size_t)(NHID / 2) * NCPAD;            // dwords
    unsigned* w1hi = (unsigned*)(ws + PK);
    unsigned* w1lo = w1hi + w1n;
    unsigned* w2hi = w1lo + w1n;
    unsigned* w2lo = w2hi + w2n;

    const dim3 blk(256);

    // 0) pack weights into WMMA-B split-bf16 layout (tiny; runs every call)
    gcn_pack_w<<<((int)w1n + 255) / 256, blk, 0, stream>>>(W1, w1hi, w1lo, NFEAT, NHID, NHID);
    gcn_pack_w<<<((int)w2n + 255) / 256, blk, 0, stream>>>(W2, w2hi, w2lo, NHID, NCPAD, NCLASS);

    // 1) sup1 = x @ W1          (one wave per 16x64 output block)
    const int waves1 = (N / 16) * (NHID / 64);
    gcn_gemm1<<<(waves1 * 32 + 255) / 256, blk, 0, stream>>>(x, w1hi, w1lo, sup1, N);

    // 2) h1 = broadcast(b1); h1[dst] += ew * sup1[src]
    gcn_init_bias<<<(N * NHID + 255) / 256, blk, 0, stream>>>(h1, b1, N, NHID);
    gcn_scatter1<<<(E * 64 + 255) / 256, blk, 0, stream>>>(sup1, src, dst, ew, h1, E);

    // 3) sup2 = relu(h1) @ W2   (one wave per 16-row stripe; sup1 region dead)
    const int waves2 = N / 16;
    gcn_gemm2<<<(waves2 * 32 + 255) / 256, blk, 0, stream>>>(h1, w2hi, w2lo, sup2, N);

    // 4) agg2 = broadcast(b2); agg2[dst] += ew * sup2[src]
    gcn_init_bias<<<(N * NCLASS + 255) / 256, blk, 0, stream>>>(agg2, b2, N, NCLASS);
    gcn_scatter2<<<(E * 32 + 255) / 256, blk, 0, stream>>>(sup2, src, dst, ew, agg2, E);

    // 5) out = log_softmax(agg2, axis=1)
    gcn_logsoftmax<<<(N * 32 + 255) / 256, blk, 0, stream>>>(agg2, out, N);
}
